// DVGCL_19155554140395
// MI455X (gfx1250) — compile-verified, matrix operand order checked by reading, not run on hardware
//
#include <hip/hip_runtime.h>
#include <hip/hip_bf16.h>
#include <math.h>

// ---------------- constants (from reference) ----------------
#define N_USERS   50000
#define N_ITEMS   100000
#define NTOT      150000          // N_USERS + N_ITEMS
#define DD        64
#define T_SIZE    32
#define N_INT     128
#define N_LAYERS  3
#define BB        4096
#define TEMP      0.2f
#define KL_REG    0.01f
#define EMB_REG   1e-5f
#define INT_REG   1e-5f
#define SSL_REG   0.1f

typedef _Float16 v16h __attribute__((ext_vector_type(16)));
typedef _Float16 h8   __attribute__((ext_vector_type(8)));
typedef float    v8f  __attribute__((ext_vector_type(8)));

__device__ __forceinline__ float softplusf(float x) {
    return (x > 20.f) ? x : log1pf(__expf(x));
}

// ---------------- init: cur = acc = concat(user_emb, item_emb) ----------------
__global__ void dvgcl_init(const float* __restrict__ ue, const float* __restrict__ ie,
                           float* __restrict__ cur, float* __restrict__ acc) {
    int i4 = blockIdx.x * blockDim.x + threadIdx.x;       // float4 index
    if (i4 >= NTOT * (DD / 4)) return;
    int row = i4 >> 4;                                     // 16 float4 per row
    int col = i4 & 15;
    const float4* src = (row < N_USERS)
        ? reinterpret_cast<const float4*>(ue) + row * 16
        : reinterpret_cast<const float4*>(ie) + (row - N_USERS) * 16;
    float4 v = src[col];
    reinterpret_cast<float4*>(cur)[i4] = v;
    reinterpret_cast<float4*>(acc)[i4] = v;
}

// ---------------- degree + d^-1/2 ----------------
__global__ void dvgcl_deg(const int* __restrict__ h, float* __restrict__ deg, int E) {
    int i = blockIdx.x * blockDim.x + threadIdx.x;
    if (i < E) atomicAdd(&deg[h[i]], 1.0f);
}
__global__ void dvgcl_rsqrt(float* __restrict__ deg) {
    int i = blockIdx.x * blockDim.x + threadIdx.x;
    if (i < NTOT) deg[i] = rsqrtf(deg[i]);
}

// ---------------- one propagation layer: nxt[h] += g * cur[t] ----------------
// one wave per edge, lane = dim (2 dims/lane), coalesced gather + atomics
__global__ void dvgcl_prop(const int* __restrict__ h, const int* __restrict__ t,
                           const float* __restrict__ dinv,
                           const float* __restrict__ cur, float* __restrict__ nxt, int E) {
    int wave = (blockIdx.x * blockDim.x + threadIdx.x) >> 5;
    int lane = threadIdx.x & 31;
    if (wave >= E) return;
    int hi = h[wave], ti = t[wave];
    float g = dinv[hi] * dinv[ti];
    float a = cur[ti * DD + lane];
    float b = cur[ti * DD + lane + 32];
    atomicAdd(&nxt[hi * DD + lane],      g * a);
    atomicAdd(&nxt[hi * DD + lane + 32], g * b);
}

__global__ void dvgcl_accadd(float* __restrict__ acc, const float* __restrict__ nxt, int n4) {
    int i = blockIdx.x * blockDim.x + threadIdx.x;
    if (i >= n4) return;
    float4 a = reinterpret_cast<float4*>(acc)[i];
    float4 b = reinterpret_cast<const float4*>(nxt)[i];
    a.x += b.x; a.y += b.y; a.z += b.z; a.w += b.w;
    reinterpret_cast<float4*>(acc)[i] = a;
}

// ---------------- KL over all N rows (std recomputed on the fly) ----------------
__global__ void dvgcl_kl(const float* __restrict__ acc, const float* __restrict__ lin_w,
                         const float* __restrict__ lin_b, float* __restrict__ accum) {
    __shared__ float sp[T_SIZE];
    __shared__ float red[DD];
    int row = blockIdx.x;
    int d = threadIdx.x;                   // 64 threads
    const float* r = acc + row * DD;
    if (d < T_SIZE) sp[d] = softplusf(r[d]);
    __syncthreads();
    float s = lin_b[d] + 1e-8f;
    #pragma unroll
    for (int tt = 0; tt < T_SIZE; ++tt) s += sp[tt] * lin_w[d * T_SIZE + tt];
    float mean = r[d];
    float kl = -0.5f * (1.0f + 2.0f * s - mean * mean - __expf(2.0f * s));
    if (!isfinite(kl)) kl = 0.0f;
    red[d] = kl; __syncthreads();
    for (int st = 32; st > 0; st >>= 1) { if (d < st) red[d] += red[d + st]; __syncthreads(); }
    if (d == 0) atomicAdd(&accum[1], red[0]);
}

// ---------------- BPR ----------------
__global__ void dvgcl_bpr(const float* __restrict__ acc, const int* __restrict__ users,
                          const int* __restrict__ pos, const int* __restrict__ neg,
                          float* __restrict__ accum) {
    int b = blockIdx.x * blockDim.x + threadIdx.x;
    if (b >= BB) return;
    const float* u = acc + users[b] * DD;
    const float* p = acc + (N_USERS + pos[b]) * DD;
    const float* n = acc + (N_USERS + neg[b]) * DD;
    float ps = 0.f, ns = 0.f;
    #pragma unroll 8
    for (int d = 0; d < DD; ++d) { ps += u[d] * p[d]; ns += u[d] * n[d]; }
    atomicAdd(&accum[0], softplusf(ns - ps));
}

// ---------------- embedding / intent regularizers ----------------
__global__ void dvgcl_embreg(const float* __restrict__ ue, const float* __restrict__ ie,
                             const int* __restrict__ users, const int* __restrict__ pos,
                             const int* __restrict__ neg, float* __restrict__ accum) {
    int i = blockIdx.x * blockDim.x + threadIdx.x;
    if (i >= 3 * BB) return;
    int which = i / BB, b = i % BB;
    const float* r = (which == 0) ? ue + users[b] * DD
                   : (which == 1) ? ie + pos[b] * DD
                                  : ie + neg[b] * DD;
    float s = 0.f;
    #pragma unroll 8
    for (int d = 0; d < DD; ++d) s += r[d] * r[d];
    atomicAdd(&accum[2], s);
}
__global__ void dvgcl_intreg(const float* __restrict__ ui, const float* __restrict__ ii,
                             float* __restrict__ accum) {
    __shared__ float red[256];
    int tid = threadIdx.x;
    int total = 2 * DD * N_INT;
    float s = 0.f;
    for (int i = blockIdx.x * blockDim.x + tid; i < total; i += gridDim.x * blockDim.x) {
        float v = (i < DD * N_INT) ? ui[i] : ii[i - DD * N_INT];
        s += v * v;
    }
    red[tid] = s; __syncthreads();
    for (int st = 128; st > 0; st >>= 1) { if (tid < st) red[tid] += red[tid + st]; __syncthreads(); }
    if (tid == 0) atomicAdd(&accum[3], red[0]);
}

// ---------------- intent projection at gathered rows, normalized, f32+f16 ----------------
__global__ void dvgcl_intent(const float* __restrict__ acc, const int* __restrict__ idx,
                             int row_off, const float* __restrict__ W,
                             float* __restrict__ outf, _Float16* __restrict__ outh) {
    __shared__ float srow[DD];
    __shared__ float lg[N_INT];
    __shared__ float red[N_INT];
    int b = blockIdx.x;
    int tid = threadIdx.x;                 // 128 threads
    int row = row_off + idx[b];
    if (tid < DD) srow[tid] = acc[row * DD + tid];
    __syncthreads();
    float l = 0.f;
    #pragma unroll 8
    for (int d = 0; d < DD; ++d) l += srow[d] * W[d * N_INT + tid];
    red[tid] = l; __syncthreads();
    for (int s = 64; s > 0; s >>= 1) { if (tid < s) red[tid] = fmaxf(red[tid], red[tid + s]); __syncthreads(); }
    float mx = red[0]; __syncthreads();
    float p = __expf(l - mx);
    red[tid] = p; __syncthreads();
    for (int s = 64; s > 0; s >>= 1) { if (tid < s) red[tid] += red[tid + s]; __syncthreads(); }
    float psum = red[0]; __syncthreads();
    lg[tid] = p / psum; __syncthreads();
    float o = 0.f;
    if (tid < DD) {
        #pragma unroll 8
        for (int k = 0; k < N_INT; ++k) o += lg[k] * W[tid * N_INT + k];
    }
    red[tid] = (tid < DD) ? o * o : 0.f; __syncthreads();
    for (int s = 64; s > 0; s >>= 1) { if (tid < s) red[tid] += red[tid + s]; __syncthreads(); }
    float nrm = sqrtf(red[0]);
    if (tid < DD) {
        float v = o / nrm;
        outf[b * DD + tid] = v;
        outh[b * DD + tid] = (_Float16)v;
    }
}

// ---------------- gen_emb at gathered rows, normalized, f32+f16 ----------------
__global__ void dvgcl_gen(const float* __restrict__ acc, const float* __restrict__ eps,
                          const float* __restrict__ lin_w, const float* __restrict__ lin_b,
                          const int* __restrict__ idx, int row_off,
                          float* __restrict__ outf, _Float16* __restrict__ outh) {
    __shared__ float sp[T_SIZE];
    __shared__ float red[DD];
    int b = blockIdx.x;
    int d = threadIdx.x;                   // 64 threads
    int row = row_off + idx[b];
    const float* r = acc + row * DD;
    if (d < T_SIZE) sp[d] = softplusf(r[d]);
    __syncthreads();
    float s = lin_b[d] + 1e-8f;
    #pragma unroll
    for (int tt = 0; tt < T_SIZE; ++tt) s += sp[tt] * lin_w[d * T_SIZE + tt];
    float g = r[d] + eps[row * DD + d] * s;
    red[d] = g * g; __syncthreads();
    for (int st = 32; st > 0; st >>= 1) { if (d < st) red[d] += red[d + st]; __syncthreads(); }
    float nrm = sqrtf(red[0]);
    float v = g / nrm;
    outf[b * DD + d] = v;
    outh[b * DD + d] = (_Float16)v;
}

// ---------------- InfoNCE positive terms ----------------
__global__ void dvgcl_pos(const float* __restrict__ ug, const float* __restrict__ ui,
                          const float* __restrict__ ig, const float* __restrict__ ii,
                          float* __restrict__ pos_u, float* __restrict__ pos_i) {
    int b = blockIdx.x * blockDim.x + threadIdx.x;
    if (b >= BB) return;
    float s1 = 0.f, s2 = 0.f;
    #pragma unroll 8
    for (int d = 0; d < DD; ++d) {
        s1 += ug[b * DD + d] * ui[b * DD + d];
        s2 += ig[b * DD + d] * ii[b * DD + d];
    }
    pos_u[b] = __expf(s1 / TEMP);
    pos_i[b] = __expf(s2 / TEMP);
}

// ---------------- InfoNCE denominators via WMMA f16 (f32 accumulate) ----------------
// One wave owns one 16-row tile of e1 (A fragments loaded once from global) and
// loops over all 256 column tiles of e2. The e2 tiles are staged into LDS once
// per 8-wave workgroup with async global->LDS copies (ASYNCcnt), double-buffered
// so the next tile streams in while WMMA consumes the current one from LDS.
__device__ __forceinline__ v16h dvgcl_frag(const _Float16* __restrict__ p, int row,
                                           int hq, int koff) {
    const _Float16* base = p + row * DD + hq * 8 + koff;
    h8 lo = *reinterpret_cast<const h8*>(base);
    h8 hi = *reinterpret_cast<const h8*>(base + 16);
    v16h a;
    #pragma unroll
    for (int i = 0; i < 8; ++i) { a[i] = lo[i]; a[i + 8] = hi[i]; }
    return a;
}

__device__ __forceinline__ void dvgcl_async_tile(const _Float16* __restrict__ e2,
                                                 const _Float16* lds_buf,
                                                 int jtile, int tid) {
    // 256 threads x 8B (4 halfs) = 2KB = one 16x64 f16 tile
    uint32_t lds = (uint32_t)(uintptr_t)lds_buf + (uint32_t)tid * 8u;
    const _Float16* g = e2 + (jtile * 16 + (tid >> 4)) * DD + (tid & 15) * 4;
    asm volatile("global_load_async_to_lds_b64 %0, %1, off"
                 :: "v"(lds), "v"((uint64_t)(uintptr_t)g)
                 : "memory");
}

__global__ void dvgcl_neg(const _Float16* __restrict__ e1, const _Float16* __restrict__ e2,
                          float* __restrict__ neg) {
    __shared__ __align__(16) _Float16 tile[2][16 * DD];   // double-buffered 2KB tiles
    int tid  = threadIdx.x;
    int lane = tid & 31;
    int wid  = tid >> 5;
    int m0 = (blockIdx.x * 8 + wid) * 16;
    int hq = lane >> 4;
    int mrow = m0 + (lane & 15);

    v16h a0 = dvgcl_frag(e1, mrow, hq, 0);
    v16h a1 = dvgcl_frag(e1, mrow, hq, 32);

    // prologue: stage tile 0
    dvgcl_async_tile(e2, &tile[0][0], 0, tid);
    asm volatile("s_wait_asynccnt 0x0" ::: "memory");
    __syncthreads();

    float rs[8];
    #pragma unroll
    for (int r = 0; r < 8; ++r) rs[r] = 0.f;
    const float invT = 1.0f / TEMP;

    for (int j = 0; j < BB / 16; ++j) {
        int buf = j & 1;
        if (j + 1 < BB / 16)
            dvgcl_async_tile(e2, &tile[buf ^ 1][0], j + 1, tid);

        // B fragments from LDS (ds_load), ISA 16-bit layout: lane = N%16
        const _Float16* base = &tile[buf][0] + (lane & 15) * DD + hq * 8;
        h8 lo0 = *reinterpret_cast<const h8*>(base);
        h8 hi0 = *reinterpret_cast<const h8*>(base + 16);
        h8 lo1 = *reinterpret_cast<const h8*>(base + 32);
        h8 hi1 = *reinterpret_cast<const h8*>(base + 48);
        v16h b0, b1;
        #pragma unroll
        for (int i = 0; i < 8; ++i) {
            b0[i] = lo0[i]; b0[i + 8] = hi0[i];
            b1[i] = lo1[i]; b1[i + 8] = hi1[i];
        }

        v8f c = {};
        c = __builtin_amdgcn_wmma_f32_16x16x32_f16(false, a0, false, b0, (short)0, c, false, false);
        c = __builtin_amdgcn_wmma_f32_16x16x32_f16(false, a1, false, b1, (short)0, c, false, false);
        #pragma unroll
        for (int r = 0; r < 8; ++r) rs[r] += __expf(c[r] * invT);

        // next buffer landed + everyone done reading current before overwrite
        asm volatile("s_wait_asynccnt 0x0" ::: "memory");
        __syncthreads();
    }

    #pragma unroll
    for (int r = 0; r < 8; ++r) {
        float v = rs[r];
        v += __shfl_xor(v, 1, 32);
        v += __shfl_xor(v, 2, 32);
        v += __shfl_xor(v, 4, 32);
        v += __shfl_xor(v, 8, 32);
        if ((lane & 15) == 0) neg[m0 + hq * 8 + r] = v;
    }
}

// ---------------- final: assemble the four losses ----------------
__global__ void dvgcl_final(const float* __restrict__ pos_u, const float* __restrict__ neg_u,
                            const float* __restrict__ pos_i, const float* __restrict__ neg_i,
                            const float* __restrict__ accum, float* __restrict__ out) {
    __shared__ float r1[256], r2[256];
    int tid = threadIdx.x;
    float t1 = 0.f, t2 = 0.f;
    for (int b = tid; b < BB; b += 256) {
        t1 += -logf(pos_u[b] / (neg_u[b] + 1e-8f) + 1e-8f);
        t2 += -logf(pos_i[b] / (neg_i[b] + 1e-8f) + 1e-8f);
    }
    r1[tid] = t1; r2[tid] = t2; __syncthreads();
    for (int s = 128; s > 0; s >>= 1) {
        if (tid < s) { r1[tid] += r1[tid + s]; r2[tid] += r2[tid + s]; }
        __syncthreads();
    }
    if (tid == 0) {
        out[0] = accum[0] / (float)BB + KL_REG * (accum[1] / (float)NTOT);   // gen_loss
        out[1] = SSL_REG * (r1[0] / (float)BB + r2[0] / (float)BB);          // cl_loss
        out[2] = EMB_REG * accum[2];                                         // emb_loss
        out[3] = INT_REG * accum[3];                                         // int_loss
    }
}

// ---------------- launcher ----------------
extern "C" void kernel_launch(void* const* d_in, const int* in_sizes, int n_in,
                              void* d_out, int out_size, void* d_ws, size_t ws_size,
                              hipStream_t stream) {
    const float* user_emb    = (const float*)d_in[0];
    const float* item_emb    = (const float*)d_in[1];
    const float* user_intent = (const float*)d_in[2];
    const float* item_intent = (const float*)d_in[3];
    const float* lin_w       = (const float*)d_in[4];
    const float* lin_b       = (const float*)d_in[5];
    const float* eps         = (const float*)d_in[6];
    const int*   h_list      = (const int*)d_in[7];
    const int*   t_list      = (const int*)d_in[8];
    const int*   users       = (const int*)d_in[9];
    const int*   pos_items   = (const int*)d_in[10];
    const int*   neg_items   = (const int*)d_in[11];
    const int E = in_sizes[7];
    float* out = (float*)d_out;

    // workspace carve-up (256B-aligned chunks)
    char* w = (char*)d_ws;
    size_t o = 0;
    auto carve = [&](size_t bytes) -> char* {
        char* p = w + o;
        o += (bytes + 255) & ~(size_t)255;
        return p;
    };
    const size_t NF = (size_t)NTOT * DD;
    float* cur   = (float*)carve(NF * 4);
    float* nxt   = (float*)carve(NF * 4);
    float* acc   = (float*)carve(NF * 4);
    float* dinv  = (float*)carve(NTOT * 4);          // deg, then rsqrt in place
    float* u_gen = (float*)carve(BB * DD * 4);
    float* i_gen = (float*)carve(BB * DD * 4);
    float* u_int = (float*)carve(BB * DD * 4);
    float* i_int = (float*)carve(BB * DD * 4);
    _Float16* u_gen_h = (_Float16*)carve(BB * DD * 2);
    _Float16* i_gen_h = (_Float16*)carve(BB * DD * 2);
    _Float16* u_int_h = (_Float16*)carve(BB * DD * 2);
    _Float16* i_int_h = (_Float16*)carve(BB * DD * 2);
    float* pos_u = (float*)carve(BB * 4);
    float* pos_i = (float*)carve(BB * 4);
    float* neg_u = (float*)carve(BB * 4);
    float* neg_i = (float*)carve(BB * 4);
    float* accum = (float*)carve(16 * 4);            // [bpr, kl, emb, int]

    hipMemsetAsync(dinv, 0, NTOT * 4, stream);
    hipMemsetAsync(accum, 0, 16 * 4, stream);

    // ego embedding + accumulator
    dvgcl_init<<<(NTOT * (DD / 4) + 255) / 256, 256, 0, stream>>>(user_emb, item_emb, cur, acc);

    // degrees
    dvgcl_deg<<<(E + 255) / 256, 256, 0, stream>>>(h_list, dinv, E);
    dvgcl_rsqrt<<<(NTOT + 255) / 256, 256, 0, stream>>>(dinv);

    // 3 propagation layers (ping-pong cur/nxt)
    for (int l = 0; l < N_LAYERS; ++l) {
        hipMemsetAsync(nxt, 0, NF * 4, stream);
        dvgcl_prop<<<(E + 7) / 8, 256, 0, stream>>>(h_list, t_list, dinv, cur, nxt, E);
        dvgcl_accadd<<<(int)((NF / 4 + 255) / 256), 256, 0, stream>>>(acc, nxt, (int)(NF / 4));
        float* tmp = cur; cur = nxt; nxt = tmp;
    }

    // losses over acc (= all_emb)
    dvgcl_kl<<<NTOT, DD, 0, stream>>>(acc, lin_w, lin_b, accum);
    dvgcl_bpr<<<(BB + 255) / 256, 256, 0, stream>>>(acc, users, pos_items, neg_items, accum);
    dvgcl_embreg<<<(3 * BB + 255) / 256, 256, 0, stream>>>(user_emb, item_emb, users, pos_items, neg_items, accum);
    dvgcl_intreg<<<8, 256, 0, stream>>>(user_intent, item_intent, accum);

    // gathered intent + generated embeddings (normalized, f32 + f16 copies)
    dvgcl_intent<<<BB, N_INT, 0, stream>>>(acc, users, 0, user_intent, u_int, u_int_h);
    dvgcl_intent<<<BB, N_INT, 0, stream>>>(acc, pos_items, N_USERS, item_intent, i_int, i_int_h);
    dvgcl_gen<<<BB, DD, 0, stream>>>(acc, eps, lin_w, lin_b, users, 0, u_gen, u_gen_h);
    dvgcl_gen<<<BB, DD, 0, stream>>>(acc, eps, lin_w, lin_b, pos_items, N_USERS, i_gen, i_gen_h);

    // InfoNCE
    dvgcl_pos<<<(BB + 255) / 256, 256, 0, stream>>>(u_gen, u_int, i_gen, i_int, pos_u, pos_i);
    dvgcl_neg<<<(BB / 16) / 8, 256, 0, stream>>>(u_gen_h, u_int_h, neg_u);   // 256 waves
    dvgcl_neg<<<(BB / 16) / 8, 256, 0, stream>>>(i_gen_h, i_int_h, neg_i);

    dvgcl_final<<<1, 256, 0, stream>>>(pos_u, neg_u, pos_i, neg_i, accum, out);
}
